// Phi3SmallSelfAttention_58136677318743
// MI455X (gfx1250) — matrix operand order repaired and checked by reading, not compiled, not run on hardware
//
#include <hip/hip_runtime.h>
#include <cstdint>
#include <cstddef>

#define SEQ       2048
#define HIDDEN    4096
#define NHEADS    32
#define NKV       8
#define HDIM      128
#define OQKV      6144          // (32 + 2*8) * 128
#define SMSCALE   (1.0f/128.0f)
#define LOCALB    16
#define VSTRIDE   8

typedef __attribute__((ext_vector_type(16))) __bf16 v16bf;
typedef __attribute__((ext_vector_type(8)))  float  v8f;

// ---------- fp32 -> bf16 (round-to-nearest-even) ----------
static __device__ __forceinline__ unsigned short f2bf(float f) {
  unsigned int u = __float_as_uint(f);
  u += 0x7FFFu + ((u >> 16) & 1u);
  return (unsigned short)(u >> 16);
}
static __device__ __forceinline__ unsigned int pack2(float a, float b) {
  return (unsigned int)f2bf(a) | ((unsigned int)f2bf(b) << 16);
}

// ---------- CDNA5 async global->LDS copy (ASYNCcnt tracked) ----------
// LDS byte address = low 32 bits of the generic shared pointer (shared
// aperture is 2^32-aligned: ADDR[63:32] == {SHARED_BASE,16'b0}).
static __device__ __forceinline__ void async_copy_b128(void* lds_dst, const void* g_src) {
  unsigned l = (unsigned)(uintptr_t)lds_dst;
  asm volatile("global_load_async_to_lds_b128 %0, %1, off"
               :: "v"(l), "v"(g_src)
               : "memory");
}
static __device__ __forceinline__ void async_wait0() {
  asm volatile("s_wait_asynccnt 0x0" ::: "memory");
}

union FragBF { v16bf v; uint4 q[2]; };

// A fragment: 16x32 bf16, rows row-major at `base`, row stride `ldr` (elements).
// lane = m (0..15) + half*16 ; VGPR0-3 = K[half*8 .. +7], VGPR4-7 = K[16+half*8 .. +7]
static __device__ __forceinline__ v16bf frag_a(const unsigned short* base, int ldr) {
  const int lane = threadIdx.x & 31;
  const int m = lane & 15, half = lane >> 4;
  FragBF f;
  f.q[0] = *(const uint4*)(base + m * ldr + half * 8);
  f.q[1] = *(const uint4*)(base + m * ldr + 16 + half * 8);
  return f.v;
}
// B fragment: 32x16 bf16 where LDS holds B^T (n-major rows of K values).
// lane = n (0..15) + khalf*16 ; VGPR0-3 = K[khalf*8 .. +7], VGPR4-7 = K[16+khalf*8 .. +7]
static __device__ __forceinline__ v16bf frag_b(const unsigned short* base, int ldr) {
  const int lane = threadIdx.x & 31;
  const int n = lane & 15, kh = lane >> 4;
  FragBF f;
  f.q[0] = *(const uint4*)(base + n * ldr + kh * 8);
  f.q[1] = *(const uint4*)(base + n * ldr + 16 + kh * 8);
  return f.v;
}

// =====================================================================
// GEMM: C[M,N] = X[M,K] * W[N,K]^T + bias[N]   (fp32 in, bf16 WMMA, fp32 out)
// Block tile 128x128, 8 waves each 32x64, K-step 64 (16 WMMA / barrier pair).
// =====================================================================
__global__ __launch_bounds__(256) void gemm_xwT_bias_kernel(
    const float* __restrict__ X, const float* __restrict__ W,
    const float* __restrict__ bias, float* __restrict__ C,
    int M, int N, int K)
{
  __shared__ unsigned short As[128 * 72];   // 128 rows x 64 K, stride 72 (144B rows, 16B aligned)
  __shared__ unsigned short Bs[128 * 72];

  const int tid  = threadIdx.x;
  const int wave = tid >> 5;
  const int lane = tid & 31;
  const int m0 = blockIdx.y * 128;
  const int n0 = blockIdx.x * 128;
  const int wm = (wave >> 1) * 32;     // 0,32,64,96
  const int wn = (wave & 1) * 64;      // 0,64

  v8f acc[2][4] = {};

  for (int k0 = 0; k0 < K; k0 += 64) {
    // Stage 128x64 fp32 of X and W, converting to bf16 in-register.
    #pragma unroll
    for (int i = 0; i < 8; ++i) {
      int idx = tid + i * 256;              // 2048 float4 slots per matrix
      int r = idx >> 4, c4 = (idx & 15) * 4;
      float4 a = *(const float4*)(X + (size_t)(m0 + r) * K + k0 + c4);
      unsigned int* da = (unsigned int*)(As + r * 72 + c4);
      da[0] = pack2(a.x, a.y);
      da[1] = pack2(a.z, a.w);
      float4 b = *(const float4*)(W + (size_t)(n0 + r) * K + k0 + c4);
      unsigned int* db = (unsigned int*)(Bs + r * 72 + c4);
      db[0] = pack2(b.x, b.y);
      db[1] = pack2(b.z, b.w);
      if ((idx & 3) == 0) {   // one prefetch per 2 cachelines of next tile
        __builtin_prefetch(X + (size_t)(m0 + r) * K + k0 + c4 + 64, 0, 1);
        __builtin_prefetch(W + (size_t)(n0 + r) * K + k0 + c4 + 64, 0, 1);
      }
    }
    __syncthreads();

    #pragma unroll
    for (int ks = 0; ks < 2; ++ks) {
      v16bf af[2], bfv[4];
      #pragma unroll
      for (int i = 0; i < 2; ++i) af[i]  = frag_a(As + (wm + i * 16) * 72 + ks * 32, 72);
      #pragma unroll
      for (int j = 0; j < 4; ++j) bfv[j] = frag_b(Bs + (wn + j * 16) * 72 + ks * 32, 72);
      #pragma unroll
      for (int i = 0; i < 2; ++i)
        #pragma unroll
        for (int j = 0; j < 4; ++j)
          acc[i][j] = __builtin_amdgcn_wmma_f32_16x16x32_bf16(
              false, af[i], false, bfv[j], (short)0, acc[i][j], false, false);
    }
    __syncthreads();
  }

  // Store D: lane holds col n = lane&15, rows (lane>>4)*8 + r
  const int n16 = lane & 15;
  const int ro  = (lane >> 4) * 8;
  #pragma unroll
  for (int i = 0; i < 2; ++i)
    for (int j = 0; j < 4; ++j) {
      int cn = n0 + wn + j * 16 + n16;
      float bv = bias[cn];
      #pragma unroll
      for (int r = 0; r < 8; ++r) {
        int cm = m0 + wm + i * 16 + ro + r;
        C[(size_t)cm * N + cn] = acc[i][j][r] + bv;
      }
    }
}

// =====================================================================
// RoPE + split qkv[S,6144] fp32 -> q_bf[32][S][128], k_bf[8][S][128],
// v_bf[8][128][S] (V stored transposed for WMMA B-fragment loads).
// =====================================================================
__global__ __launch_bounds__(256) void rope_split_kernel(
    const float* __restrict__ qkv,
    unsigned short* __restrict__ q_bf,
    unsigned short* __restrict__ k_bf,
    unsigned short* __restrict__ v_bf)
{
  const int s = blockIdx.x;
  const float* row = qkv + (size_t)s * OQKV;
  const float NEG_LN_BASE = -13.815510558f;   // -ln(1e6)

  for (int idx = threadIdx.x; idx < OQKV; idx += blockDim.x) {
    int kvh = idx / 768;
    int rem = idx - kvh * 768;
    int g = rem >> 7;          // 0..3 q, 4 k, 5 v
    int d = rem & 127;
    float x = row[idx];
    if (g < 5) {
      int j = d & 63;
      float invf = __expf((float)(2 * j) * (1.0f / 128.0f) * NEG_LN_BASE);
      float ang = (float)s * invf;             // POS_SCALE = 1.0
      float sn, cs;
      __sincosf(ang, &sn, &cs);
      float partner = (d < 64) ? row[idx + 64] : row[idx - 64];
      float val = (d < 64) ? (x * cs - partner * sn) : (x * cs + partner * sn);
      if (g < 4) {
        int head = kvh * 4 + g;
        q_bf[((size_t)head * SEQ + s) * HDIM + d] = f2bf(val);
      } else {
        k_bf[((size_t)kvh * SEQ + s) * HDIM + d] = f2bf(val);
      }
    } else {
      v_bf[((size_t)kvh * HDIM + d) * SEQ + s] = f2bf(x);   // transposed store
    }
  }
}

// =====================================================================
// Blocksparse flash attention. Grid: (S/64 query blocks, 32 heads).
// Tiles staged with GLOBAL_LOAD_ASYNC_TO_LDS_B128 (ASYNCcnt).
// K-tile and V^T-tile share one LDS buffer (phase separated).
// =====================================================================
__global__ __launch_bounds__(256) void attn_blocksparse_kernel(
    const unsigned short* __restrict__ q_bf,
    const unsigned short* __restrict__ k_bf,
    const unsigned short* __restrict__ v_bf,
    float* __restrict__ ctx)
{
  __shared__ unsigned short Qs[64 * 128];    // 16 KB, persistent
  __shared__ unsigned short KVs[128 * 64];   // 16 KB: K[64][128] then V^T[128][64]
  __shared__ float          Ss[64 * 64];     // 16 KB raw scores
  __shared__ unsigned short Ps[64 * 64];     //  8 KB bf16 probabilities
  __shared__ float m_s[64], l_s[64], c_s[64];

  const int tid  = threadIdx.x;
  const int lane = tid & 31;
  const int wave = tid >> 5;
  const int qb = blockIdx.x;
  const int h  = blockIdx.y;
  const int kv = h >> 2;                     // GQA: 4 q heads per kv head

  // Async-load Q tile 64x128 bf16 (waited together with first K tile)
  {
    const unsigned short* qbase = q_bf + ((size_t)h * SEQ + (size_t)qb * 64) * HDIM;
    #pragma unroll
    for (int i = 0; i < 4; ++i) {
      int idx = tid + i * 256;               // 1024 b128 transfers
      int r = idx >> 4, c = (idx & 15) * 8;
      async_copy_b128(Qs + r * 128 + c, qbase + (size_t)r * HDIM + c);
    }
  }
  if (tid < 64) { m_s[tid] = -3.0e38f; l_s[tid] = 0.0f; }

  v8f oacc[4] = {};
  const int tr  = wave >> 1;                 // output row tile (0..3)
  const int cg  = wave & 1;                  // output col group (64 dims each)
  const int n16 = lane & 15;
  const int ro  = (lane >> 4) * 8;

  for (int jb = 0; jb <= qb; ++jb) {
    bool allowed = (qb - jb < LOCALB) || (((jb + 1) % VSTRIDE) == 0);
    if (!allowed) continue;                  // uniform across block

    __syncthreads();                         // prior PV done with KVs
    {                                        // K tile 64x128 (async)
      const unsigned short* kbase = k_bf + ((size_t)kv * SEQ + (size_t)jb * 64) * HDIM;
      #pragma unroll
      for (int i = 0; i < 4; ++i) {
        int idx = tid + i * 256;
        int r = idx >> 4, c = (idx & 15) * 8;
        async_copy_b128(KVs + r * 128 + c, kbase + (size_t)r * HDIM + c);
      }
    }
    async_wait0();                           // this wave's Q+K copies landed
    __syncthreads();                         // everyone's copies landed

    // Phase A: S = (Q K^T) * scale, causal mask on diagonal block. 2 tiles/wave.
    #pragma unroll
    for (int t2 = 0; t2 < 2; ++t2) {
      int t = wave * 2 + t2;
      int str = t >> 2, stc = t & 3;
      v8f sacc = {};
      #pragma unroll
      for (int ks = 0; ks < 4; ++ks) {
        v16bf a = frag_a(Qs  + (str * 16) * 128 + ks * 32, 128);
        v16bf b = frag_b(KVs + (stc * 16) * 128 + ks * 32, 128);
        sacc = __builtin_amdgcn_wmma_f32_16x16x32_bf16(
            false, a, false, b, (short)0, sacc, false, false);
      }
      #pragma unroll
      for (int r = 0; r < 8; ++r) {
        int mrow = str * 16 + ro + r;
        int ncol = stc * 16 + n16;
        float val = sacc[r] * SMSCALE;
        if (jb == qb && ncol > mrow) val = -3.0e38f;
        Ss[mrow * 64 + ncol] = val;
      }
    }
    __syncthreads();

    // Online softmax: one thread per query row
    if (tid < 64) {
      int row = tid;
      float mold = m_s[row];
      float mx = mold;
      for (int c = 0; c < 64; ++c) mx = fmaxf(mx, Ss[row * 64 + c]);
      float cf = __expf(mold - mx);
      float lsum = 0.0f;
      for (int c = 0; c < 64; ++c) {
        float p = __expf(Ss[row * 64 + c] - mx);
        lsum += p;
        Ps[row * 64 + c] = f2bf(p);
      }
      l_s[row] = l_s[row] * cf + lsum;
      m_s[row] = mx;
      c_s[row] = cf;
    }
    __syncthreads();                         // Ss consumed, Ps/corr ready

    {                                        // V^T tile 128x64 into KVs (async)
      const unsigned short* vbase = v_bf + (size_t)kv * HDIM * SEQ + (size_t)jb * 64;
      #pragma unroll
      for (int i = 0; i < 4; ++i) {
        int idx = tid + i * 256;
        int r = idx >> 3, c = (idx & 7) * 8;
        async_copy_b128(KVs + r * 64 + c, vbase + (size_t)r * SEQ + c);
      }
    }
    async_wait0();
    __syncthreads();

    // Phase B: O = O*cf + P V
    float cf8[8];
    #pragma unroll
    for (int r = 0; r < 8; ++r) cf8[r] = c_s[tr * 16 + ro + r];
    #pragma unroll
    for (int j = 0; j < 4; ++j)
      #pragma unroll
      for (int r = 0; r < 8; ++r) oacc[j][r] *= cf8[r];
    #pragma unroll
    for (int ks = 0; ks < 2; ++ks) {
      v16bf a = frag_a(Ps + (tr * 16) * 64 + ks * 32, 64);
      #pragma unroll
      for (int j = 0; j < 4; ++j) {
        v16bf b = frag_b(KVs + (cg * 64 + j * 16) * 64 + ks * 32, 64);
        oacc[j] = __builtin_amdgcn_wmma_f32_16x16x32_bf16(
            false, a, false, b, (short)0, oacc[j], false, false);
      }
    }
  }

  // Finalize: ctx[s][h*128 + d] = O / l
  #pragma unroll
  for (int j = 0; j < 4; ++j)
    #pragma unroll
    for (int r = 0; r < 8; ++r) {
      int mrow = tr * 16 + ro + r;
      float inv = 1.0f / l_s[mrow];
      int s = qb * 64 + mrow;
      int d = cg * 64 + j * 16 + n16;
      ctx[(size_t)s * HIDDEN + (size_t)h * HDIM + d] = oacc[j][r] * inv;
    }
}

// =====================================================================
extern "C" void kernel_launch(void* const* d_in, const int* in_sizes, int n_in,
                              void* d_out, int out_size, void* d_ws, size_t ws_size,
                              hipStream_t stream)
{
  const float* hs      = (const float*)d_in[0];   // [1,2048,4096]
  const float* w_qkv   = (const float*)d_in[1];   // [6144,4096]
  const float* b_qkv   = (const float*)d_in[2];   // [6144]
  const float* w_dense = (const float*)d_in[3];   // [4096,4096]
  const float* b_dense = (const float*)d_in[4];   // [4096]
  float* out = (float*)d_out;                     // [1,2048,4096] fp32

  char* ws = (char*)d_ws;
  float*          qkv  = (float*)ws;           ws += (size_t)SEQ * OQKV * sizeof(float);          // 48 MB
  unsigned short* q_bf = (unsigned short*)ws;  ws += (size_t)NHEADS * SEQ * HDIM * 2;             // 16 MB
  unsigned short* k_bf = (unsigned short*)ws;  ws += (size_t)NKV * SEQ * HDIM * 2;                //  4 MB
  unsigned short* v_bf = (unsigned short*)ws;  ws += (size_t)NKV * HDIM * SEQ * 2;                //  4 MB
  float*          ctx  = (float*)ws;                                                             // 32 MB

  // 1) QKV projection
  gemm_xwT_bias_kernel<<<dim3(OQKV / 128, SEQ / 128), 256, 0, stream>>>(
      hs, w_qkv, b_qkv, qkv, SEQ, OQKV, HIDDEN);
  // 2) RoPE + split + V transpose (fp32 -> bf16)
  rope_split_kernel<<<dim3(SEQ), 256, 0, stream>>>(qkv, q_bf, k_bf, v_bf);
  // 3) Blocksparse flash attention
  attn_blocksparse_kernel<<<dim3(SEQ / 64, NHEADS), 256, 0, stream>>>(
      q_bf, k_bf, v_bf, ctx);
  // 4) Output projection
  gemm_xwT_bias_kernel<<<dim3(HIDDEN / 128, SEQ / 128), 256, 0, stream>>>(
      ctx, w_dense, b_dense, out, SEQ, HIDDEN, HIDDEN);
}
